// GINEncoder_12309376270692
// MI455X (gfx1250) — compile-verified, matrix operand order checked by reading, not run on hardware
//
#include <hip/hip_runtime.h>
#include <hip/hip_bf16.h>

typedef __attribute__((ext_vector_type(16))) __bf16 v16bf;
typedef __attribute__((ext_vector_type(8)))  float  v8f;

#define NNODES 100000
#define NEDGES 1600000
#define HDIM   64
#define NLAYER 3
#define WAVES  4   // waves (tiles) per block in the GEMM kernels

// ---------------- WMMA operand builders (ISA 7.12.2 layouts, wave32) --------

// A operand: 16x32 bf16 tile from row-major f32 [16][64] LDS buffer.
// lane L: m = L&15, k0 = (L>>4)*8; element j: i=j/2, g=i/4, w=i%4, p=j%2,
// k = kb + g*16 + k0 + 2w + p
__device__ inline v16bf make_a(const float* xs, int lane, int kb) {
  int m  = lane & 15;
  int k0 = (lane >> 4) << 3;
  v16bf a;
#pragma unroll
  for (int j = 0; j < 16; ++j) {
    int i = j >> 1, g = i >> 2, w = i & 3, p = j & 1;
    int k = kb + g * 16 + k0 + (w << 1) + p;
    a[j] = (__bf16)xs[m * HDIM + k];
  }
  return a;
}

// B operand: 32x16 bf16 tile, column n = nt*16 + (L&15), weights stored
// transposed in LDS as wt[n*64 + k] (bf16), same k mapping as A.
__device__ inline v16bf make_b(const __bf16* wt, int lane, int nt, int kb) {
  int n  = (nt << 4) + (lane & 15);
  int k0 = (lane >> 4) << 3;
  v16bf b;
#pragma unroll
  for (int j = 0; j < 16; ++j) {
    int i = j >> 1, g = i >> 2, w = i & 3, p = j & 1;
    int k = kb + g * 16 + k0 + (w << 1) + p;
    b[j] = wt[n * HDIM + k];
  }
  return b;
}

__device__ inline v8f gemm_tile(const v16bf& a0, const v16bf& a1,
                                const __bf16* wt, int lane, int nt) {
  v8f c = {};
  v16bf b0 = make_b(wt, lane, nt, 0);
  v16bf b1 = make_b(wt, lane, nt, 32);
  c = __builtin_amdgcn_wmma_f32_16x16x32_bf16(false, a0, false, b0, (short)0, c, false, false);
  c = __builtin_amdgcn_wmma_f32_16x16x32_bf16(false, a1, false, b1, (short)0, c, false, false);
  return c;
}

// ---------------- Kernel 1: node init MLP -----------------------------------
// h0 = relu(cat(emb[z], t) @ rw1 + rb1) @ rw2 + rb2
__global__ void init_nodes(const int* __restrict__ z, const float* __restrict__ t,
                           const float* __restrict__ emb,
                           const float* __restrict__ rw1, const float* __restrict__ rb1,
                           const float* __restrict__ rw2, const float* __restrict__ rb2,
                           float* __restrict__ h0, int n_tiles) {
  __shared__ __bf16 wt1[HDIM * HDIM];
  __shared__ __bf16 wt2[HDIM * HDIM];
  __shared__ float  xs[WAVES][16 * HDIM];

  // transpose + bf16-convert weights into LDS (rw1 rows 0..63 only)
  for (int idx = threadIdx.x; idx < HDIM * HDIM; idx += blockDim.x) {
    int k = idx >> 6, n = idx & 63;
    wt1[n * HDIM + k] = (__bf16)rw1[idx];
    wt2[n * HDIM + k] = (__bf16)rw2[idx];
  }
  __syncthreads();

  int wave = threadIdx.x >> 5;
  int lane = threadIdx.x & 31;
  int tile = blockIdx.x * WAVES + wave;
  if (tile >= n_tiles) return;          // wave-uniform: EXEC stays all-ones
  int base = tile * 16;
  float* x = xs[wave];

  // gather emb[z] tile (16 rows x 64 f32) into LDS, coalesced float4
#pragma unroll
  for (int it = 0; it < 8; ++it) {
    int idx = it * 32 + lane;           // 0..255 float4 slots
    int row = idx >> 4, q = idx & 15;
    int zz = z[base + row];
    float4 v = *(const float4*)&emb[(size_t)zz * HDIM + q * 4];
    *(float4*)&x[row * HDIM + q * 4] = v;
  }

  // GEMM1 + t-conditioning + bias + relu
  v16bf a0 = make_a(x, lane, 0);
  v16bf a1 = make_a(x, lane, 32);
  int nloc = lane & 15;
  int hi   = (lane >> 4) << 3;
#pragma unroll
  for (int nt = 0; nt < 4; ++nt) {
    v8f c = gemm_tile(a0, a1, wt1, lane, nt);
    int n = nt * 16 + nloc;
    float bias = rb1[n];
    float wrow = rw1[HDIM * HDIM + n];  // last row of rw1 (t weight)
#pragma unroll
    for (int r = 0; r < 8; ++r) {
      int m = hi + r;
      float v = c[r] + t[base + m] * wrow + bias;
      x[m * HDIM + n] = fmaxf(v, 0.0f); // restage y through LDS (layout swap)
    }
  }

  // GEMM2 + bias, store h0
  a0 = make_a(x, lane, 0);
  a1 = make_a(x, lane, 32);
#pragma unroll
  for (int nt = 0; nt < 4; ++nt) {
    v8f c = gemm_tile(a0, a1, wt2, lane, nt);
    int n = nt * 16 + nloc;
    float bias = rb2[n];
#pragma unroll
    for (int r = 0; r < 8; ++r)
      x[(hi + r) * HDIM + n] = c[r] + bias;
  }
#pragma unroll
  for (int it = 0; it < 8; ++it) {
    int idx = it * 32 + lane;
    int row = idx >> 4, q = idx & 15;
    *(float4*)&h0[(size_t)(base + row) * HDIM + q * 4] = *(const float4*)&x[row * HDIM + q * 4];
  }
}

// ---------------- Kernel 2: edge message + scatter-add ----------------------
// agg[dst] += relu(h[src] + edge_attr), 16 threads (float4 each) per edge
__global__ void edge_scatter(const int* __restrict__ ei, const float* __restrict__ ea,
                             const float* __restrict__ h, float* __restrict__ agg,
                             int nedges) {
  int tid = blockIdx.x * blockDim.x + threadIdx.x;
  int e = tid >> 4, q = tid & 15;
  if (e >= nedges) return;
  int src = ei[e];
  int dst = ei[nedges + e];
  float4 a = *(const float4*)&ea[(size_t)e * HDIM + q * 4];
  float4 x = *(const float4*)&h[(size_t)src * HDIM + q * 4];
  float4 v;
  v.x = fmaxf(a.x + x.x, 0.0f);
  v.y = fmaxf(a.y + x.y, 0.0f);
  v.z = fmaxf(a.z + x.z, 0.0f);
  v.w = fmaxf(a.w + x.w, 0.0f);
  float* p = &agg[(size_t)dst * HDIM + q * 4];
  atomicAdd(p + 0, v.x);
  atomicAdd(p + 1, v.y);
  atomicAdd(p + 2, v.z);
  atomicAdd(p + 3, v.w);
}

// ---------------- Kernel 3: node update (GINE MLP + residual) ---------------
// out = relu((agg+h)@w1+b1)@w2+b2 ; [relu] ; + h   (writes h in place, tile-local)
__global__ void node_update(const float* __restrict__ agg, const float* h_in,
                            const float* __restrict__ w1, const float* __restrict__ b1,
                            const float* __restrict__ w2, const float* __restrict__ b2,
                            float* h_out, int relu_out, int n_tiles) {
  __shared__ __bf16 wt1[HDIM * HDIM];
  __shared__ __bf16 wt2[HDIM * HDIM];
  __shared__ float  xs[WAVES][16 * HDIM];   // agg+h, then y, then out
  __shared__ float  hs[WAVES][16 * HDIM];   // residual

  for (int idx = threadIdx.x; idx < HDIM * HDIM; idx += blockDim.x) {
    int k = idx >> 6, n = idx & 63;
    wt1[n * HDIM + k] = (__bf16)w1[idx];
    wt2[n * HDIM + k] = (__bf16)w2[idx];
  }
  __syncthreads();

  int wave = threadIdx.x >> 5;
  int lane = threadIdx.x & 31;
  int tile = blockIdx.x * WAVES + wave;
  if (tile >= n_tiles) return;
  int base = tile * 16;
  float* x = xs[wave];
  float* hr = hs[wave];

#pragma unroll
  for (int it = 0; it < 8; ++it) {
    int idx = it * 32 + lane;
    int row = idx >> 4, q = idx & 15;
    size_t off = (size_t)(base + row) * HDIM + q * 4;
    float4 av = *(const float4*)&agg[off];
    float4 hv = *(const float4*)&h_in[off];
    float4 s;
    s.x = av.x + hv.x; s.y = av.y + hv.y; s.z = av.z + hv.z; s.w = av.w + hv.w;
    *(float4*)&x[row * HDIM + q * 4]  = s;
    *(float4*)&hr[row * HDIM + q * 4] = hv;
  }

  v16bf a0 = make_a(x, lane, 0);
  v16bf a1 = make_a(x, lane, 32);
  int nloc = lane & 15;
  int hi   = (lane >> 4) << 3;
#pragma unroll
  for (int nt = 0; nt < 4; ++nt) {
    v8f c = gemm_tile(a0, a1, wt1, lane, nt);
    int n = nt * 16 + nloc;
    float bias = b1[n];
#pragma unroll
    for (int r = 0; r < 8; ++r)
      x[(hi + r) * HDIM + n] = fmaxf(c[r] + bias, 0.0f);
  }

  a0 = make_a(x, lane, 0);
  a1 = make_a(x, lane, 32);
#pragma unroll
  for (int nt = 0; nt < 4; ++nt) {
    v8f c = gemm_tile(a0, a1, wt2, lane, nt);
    int n = nt * 16 + nloc;
    float bias = b2[n];
#pragma unroll
    for (int r = 0; r < 8; ++r) {
      int m = hi + r;
      float v = c[r] + bias;
      if (relu_out) v = fmaxf(v, 0.0f);
      x[m * HDIM + n] = v + hr[m * HDIM + n];   // residual
    }
  }
#pragma unroll
  for (int it = 0; it < 8; ++it) {
    int idx = it * 32 + lane;
    int row = idx >> 4, q = idx & 15;
    *(float4*)&h_out[(size_t)(base + row) * HDIM + q * 4] = *(const float4*)&x[row * HDIM + q * 4];
  }
}

// ---------------- Host launch -----------------------------------------------
extern "C" void kernel_launch(void* const* d_in, const int* in_sizes, int n_in,
                              void* d_out, int out_size, void* d_ws, size_t ws_size,
                              hipStream_t stream) {
  const int*   z    = (const int*)d_in[0];
  const int*   ei   = (const int*)d_in[1];
  const float* ea   = (const float*)d_in[2];
  const float* t    = (const float*)d_in[3];
  const float* emb  = (const float*)d_in[4];
  const float* rw1  = (const float*)d_in[5];
  const float* rb1  = (const float*)d_in[6];
  const float* rw2  = (const float*)d_in[7];
  const float* rb2  = (const float*)d_in[8];
  const float* cw1  = (const float*)d_in[9];
  const float* cb1  = (const float*)d_in[10];
  const float* cw2  = (const float*)d_in[11];
  const float* cb2  = (const float*)d_in[12];

  float* h   = (float*)d_out;          // node features live in d_out throughout
  float* agg = (float*)d_ws;           // scatter accumulator (N*H f32)

  const int n_tiles = NNODES / 16;     // 6250, exact
  const int gemm_blocks = (n_tiles + WAVES - 1) / WAVES;

  init_nodes<<<gemm_blocks, WAVES * 32, 0, stream>>>(z, t, emb, rw1, rb1, rw2, rb2,
                                                     h, n_tiles);

  const int edge_threads = NEDGES * 16;
  const int edge_blocks  = (edge_threads + 255) / 256;
  for (int li = 0; li < NLAYER; ++li) {
    hipMemsetAsync(agg, 0, (size_t)NNODES * HDIM * sizeof(float), stream);
    edge_scatter<<<edge_blocks, 256, 0, stream>>>(ei, ea, h, agg, NEDGES);
    node_update<<<gemm_blocks, WAVES * 32, 0, stream>>>(
        agg, h,
        cw1 + (size_t)li * HDIM * HDIM, cb1 + (size_t)li * HDIM,
        cw2 + (size_t)li * HDIM * HDIM, cb2 + (size_t)li * HDIM,
        h, (li < NLAYER - 1) ? 1 : 0, n_tiles);
  }
}
